// TaskSchedulerGNN_6786048328275
// MI455X (gfx1250) — compile-verified, matrix operand order checked by reading, not run on hardware
//
#include <hip/hip_runtime.h>
#include <hip/hip_bf16.h>
#include <math.h>

typedef __attribute__((ext_vector_type(16))) __bf16 v16bf;
typedef __attribute__((ext_vector_type(8)))  __bf16 v8bf;
typedef __attribute__((ext_vector_type(8)))  float  v8f;

#define N_NODES 100001
#define NP      100016          // N rounded up to 16
#define EDGES   3200000
#define HDIM    64
#define NM      100000          // n-1 rows scored by the MLP
#define NDAGS   1000
#define RBLK    1024            // reduction blocks

// ---------------------------------------------------------------- helpers
__device__ __forceinline__ v8bf load8(const __bf16* p) {
    return *(const v8bf*)p;
}

__device__ __forceinline__ v16bf cat16(v8bf lo, v8bf hi) {
    return __builtin_shufflevector(lo, hi, 0,1,2,3,4,5,6,7,8,9,10,11,12,13,14,15);
}

// piecewise source for the 192-wide concatenated MLP row; ks is 8-aligned and
// never crosses a 64-boundary within a chunk
__device__ __forceinline__ const __bf16* srcp(const __bf16* h, int m, int sumRow,
                                              int lastRow, int ks) {
    int region = ks >> 6;
    int row = (region == 0) ? m : ((region == 1) ? sumRow : lastRow);
    return h + (size_t)row * HDIM + (ks & 63);
}

__device__ __forceinline__ bool mask_of(const float* x, const int* dsi, int m) {
    bool isSum = false;
    if ((m % 100) == 99) {
        int k = m / 100;
        if (k < NDAGS && dsi[k] == m) isSum = true;
    }
    return (x[(size_t)m * 8 + 7] == 1.0f) && !isSum;
}

// ---------------------------------------------------------------- init / weights
__global__ void k_init_deg(float* deg) {
    int i = blockIdx.x * blockDim.x + threadIdx.x;
    if (i < NP) deg[i] = (i < N_NODES) ? 1.0f : 0.0f;   // self-loop pre-count
}

// transpose + bf16-convert W2 (64x64), M1 (192x64), M2 (64x64)
__global__ void k_cvt_weights(const float* W2, const float* M1, const float* M2,
                              __bf16* w2t, __bf16* m1t, __bf16* m2t) {
    int i = blockIdx.x * blockDim.x + threadIdx.x;
    if (i < 4096) {
        int k = i / 64, n = i % 64;
        w2t[n * 64 + k] = (__bf16)W2[i];
    } else if (i < 4096 + 12288) {
        int j = i - 4096;
        int k = j / 64, n = j % 64;
        m1t[n * 192 + k] = (__bf16)M1[j];
    } else if (i < 4096 + 12288 + 4096) {
        int j = i - 4096 - 12288;
        int k = j / 64, n = j % 64;
        m2t[n * 64 + k] = (__bf16)M2[j];
    }
}

// ---------------------------------------------------------------- degree / norm
__global__ void k_degree(const int* ei, float* deg) {
    int e = blockIdx.x * blockDim.x + threadIdx.x;
    if (e < EDGES) atomicAdd(&deg[ei[EDGES + e]], 1.0f);
}

__global__ void k_rsqrt(float* deg) {
    int i = blockIdx.x * blockDim.x + threadIdx.x;
    if (i < N_NODES) deg[i] = rsqrtf(deg[i]);           // deg >= 1 always
}

// ---------------------------------------------------------------- conv1 GEMM (K=8, scalar)
// fused epilogue: hW -> bufA, aggregation seed (b + hW*dinv^2) -> bufB
__global__ void k_gemm1(const float* __restrict__ x, const float* __restrict__ W1,
                        const float* __restrict__ dinv, const float* __restrict__ b,
                        float* __restrict__ hW, float* __restrict__ agg) {
    int t = blockIdx.x * blockDim.x + threadIdx.x;
    if (t >= N_NODES * HDIM) return;
    int r = t / HDIM, c = t % HDIM;
    float s = 0.0f;
#pragma unroll
    for (int k = 0; k < 8; ++k) s = fmaf(x[(size_t)r * 8 + k], W1[k * HDIM + c], s);
    hW[t] = s;
    float di = dinv[r];
    agg[t] = b[c] + s * di * di;
}

// ---------------------------------------------------------------- edge scatter (wave per edge)
__global__ void __launch_bounds__(256)
k_edges(const int* __restrict__ ei, const float* __restrict__ dinv,
        const float* __restrict__ hW, float* __restrict__ agg) {
    int wid  = (blockIdx.x * blockDim.x + threadIdx.x) >> 5;
    int lane = threadIdx.x & 31;
    if (wid >= EDGES) return;
    int s = ei[wid], d = ei[EDGES + wid];
    float nrm = dinv[s] * dinv[d];
    const float* hs = hW + (size_t)s * HDIM;
    float* od       = agg + (size_t)d * HDIM;
    atomicAdd(&od[lane],      hs[lane]      * nrm);
    atomicAdd(&od[lane + 32], hs[lane + 32] * nrm);
}

// ---------------------------------------------------------------- relu + bf16 convert (zero-pad)
__global__ void k_relu_cvt(const float* __restrict__ agg, __bf16* __restrict__ hbf) {
    int t = blockIdx.x * blockDim.x + threadIdx.x;
    if (t >= NP * HDIM) return;
    int r = t / HDIM;
    float v = (r < N_NODES) ? fmaxf(agg[t], 0.0f) : 0.0f;
    hbf[t] = (__bf16)v;
}

// ---------------------------------------------------------------- conv2 WMMA GEMM
// A: NP x 64 bf16; Bt: 64 x 64 bf16 (pre-transposed). Wave per 16x16 tile.
// Fused epilogue: raw hW -> outHW, aggregation seed (b + hW*dinv^2) -> agg
__global__ void __launch_bounds__(256)
k_wmma_conv2(const __bf16* __restrict__ A, const __bf16* __restrict__ Bt,
             const float* __restrict__ dinv, const float* __restrict__ bias,
             float* __restrict__ outHW, float* __restrict__ agg) {
    int wid  = (blockIdx.x * blockDim.x + threadIdx.x) >> 5;
    int lane = threadIdx.x & 31;
    const int rowTiles = NP / 16;
    if (wid >= rowTiles * 4) return;
    int colTile = wid & 3, rowTile = wid >> 2;
    int half = lane >> 4, lm = lane & 15;
    int m = rowTile * 16 + lm;
    int n = colTile * 16 + lm;

    v8f acc = {0.f, 0.f, 0.f, 0.f, 0.f, 0.f, 0.f, 0.f};
    const __bf16* arow = A  + (size_t)m * 64;
    const __bf16* brow = Bt + (size_t)n * 64;
#pragma unroll
    for (int s = 0; s < 64; s += 32) {
        int ks0 = s + 8 * half, ks1 = ks0 + 16;
        v16bf a = cat16(load8(arow + ks0), load8(arow + ks1));
        v16bf b = cat16(load8(brow + ks0), load8(brow + ks1));
        acc = __builtin_amdgcn_wmma_f32_16x16x32_bf16(false, a, false, b,
                                                      (short)0, acc, false, false);
    }
    float bv = bias[n];
#pragma unroll
    for (int r = 0; r < 8; ++r) {
        int M = rowTile * 16 + r + half * 8;
        float v = acc[r];
        size_t o = (size_t)M * HDIM + n;
        outHW[o] = v;
        if (M < N_NODES) {
            float di = dinv[M];
            agg[o] = bv + v * di * di;
        }
    }
}

// ---------------------------------------------------------------- MLP layer1: gathered K=192
__global__ void __launch_bounds__(256)
k_wmma_mlp1(const __bf16* __restrict__ h, const __bf16* __restrict__ Bt,
            const int* __restrict__ dsi, const float* __restrict__ bias,
            __bf16* __restrict__ outBf) {
    int wid  = (blockIdx.x * blockDim.x + threadIdx.x) >> 5;
    int lane = threadIdx.x & 31;
    const int rowTiles = NM / 16;                       // 6250, exact
    if (wid >= rowTiles * 4) return;
    int colTile = wid & 3, rowTile = wid >> 2;
    int half = lane >> 4, lm = lane & 15;
    int m = rowTile * 16 + lm;
    int n = colTile * 16 + lm;

    int dag = (m + 1) / 100; if (dag > NDAGS - 1) dag = NDAGS - 1;   // searchsorted-right
    int sumRow = dsi[dag];
    const int lastRow = N_NODES - 1;

    v8f acc = {0.f, 0.f, 0.f, 0.f, 0.f, 0.f, 0.f, 0.f};
    const __bf16* brow = Bt + (size_t)n * 192;
#pragma unroll
    for (int s = 0; s < 192; s += 32) {
        int ks0 = s + 8 * half, ks1 = ks0 + 16;
        v16bf a = cat16(load8(srcp(h, m, sumRow, lastRow, ks0)),
                        load8(srcp(h, m, sumRow, lastRow, ks1)));
        v16bf b = cat16(load8(brow + ks0), load8(brow + ks1));
        acc = __builtin_amdgcn_wmma_f32_16x16x32_bf16(false, a, false, b,
                                                      (short)0, acc, false, false);
    }
    float bv = bias[n];
#pragma unroll
    for (int r = 0; r < 8; ++r) {
        int M = rowTile * 16 + r + half * 8;
        float v = fmaxf(acc[r] + bv, 0.0f);
        outBf[(size_t)M * HDIM + n] = (__bf16)v;
    }
}

// ---------------------------------------------------------------- MLP layer2 + M3 dot, fused
// Wave per 16-row strip: 4 col-tiles x 2 K-steps = 8 WMMAs, A frags reused.
// z2 = relu(z1 @ M2 + mb2) stays in registers; scores = z2 . M3 + mb3 via
// in-half shfl_xor reduction (rows 0-7 live in lanes 0-15, rows 8-15 in 16-31).
__global__ void __launch_bounds__(256)
k_wmma_mlp2(const __bf16* __restrict__ A, const __bf16* __restrict__ Bt,
            const float* __restrict__ bias, const float* __restrict__ M3,
            const float* __restrict__ mb3, float* __restrict__ scores) {
    int wid  = (blockIdx.x * blockDim.x + threadIdx.x) >> 5;
    int lane = threadIdx.x & 31;
    const int rowTiles = NM / 16;                       // 6250
    if (wid >= rowTiles) return;
    int half = lane >> 4, lm = lane & 15;
    int m = wid * 16 + lm;

    const __bf16* arow = A + (size_t)m * 64;
    v16bf a0 = cat16(load8(arow + 8 * half),      load8(arow + 8 * half + 16));
    v16bf a1 = cat16(load8(arow + 32 + 8 * half), load8(arow + 32 + 8 * half + 16));

    float rowdot[8];
#pragma unroll
    for (int r = 0; r < 8; ++r) rowdot[r] = 0.0f;

#pragma unroll
    for (int t = 0; t < 4; ++t) {
        int n = t * 16 + lm;
        const __bf16* brow = Bt + (size_t)n * 64;
        v16bf b0 = cat16(load8(brow + 8 * half),      load8(brow + 8 * half + 16));
        v16bf b1 = cat16(load8(brow + 32 + 8 * half), load8(brow + 32 + 8 * half + 16));
        v8f acc = {0.f, 0.f, 0.f, 0.f, 0.f, 0.f, 0.f, 0.f};
        acc = __builtin_amdgcn_wmma_f32_16x16x32_bf16(false, a0, false, b0,
                                                      (short)0, acc, false, false);
        acc = __builtin_amdgcn_wmma_f32_16x16x32_bf16(false, a1, false, b1,
                                                      (short)0, acc, false, false);
        float bv = bias[n];
        float w3 = M3[n];
#pragma unroll
        for (int r = 0; r < 8; ++r) {
            float v = fmaxf(acc[r] + bv, 0.0f);
            rowdot[r] = fmaf(v, w3, rowdot[r]);
        }
    }
    float mb = mb3[0];
#pragma unroll
    for (int r = 0; r < 8; ++r) {
        float v = rowdot[r];
        v += __shfl_xor(v, 1, 32);
        v += __shfl_xor(v, 2, 32);
        v += __shfl_xor(v, 4, 32);
        v += __shfl_xor(v, 8, 32);                     // full 16-lane half summed
        if (lm == 0) scores[wid * 16 + r + half * 8] = v + mb;
    }
}

// ---------------------------------------------------------------- masked max/argmax (2-stage)
__global__ void k_red_max(const float* __restrict__ scores, const float* __restrict__ x,
                          const int* __restrict__ dsi, float* pmax, int* pidx) {
    __shared__ float sv[256];
    __shared__ int   si[256];
    float best = -INFINITY; int bi = 0x7fffffff;
    for (int i = blockIdx.x * blockDim.x + threadIdx.x; i < NM; i += gridDim.x * blockDim.x) {
        if (mask_of(x, dsi, i)) {
            float v = scores[i];
            if (v > best || (v == best && i < bi)) { best = v; bi = i; }
        }
    }
    sv[threadIdx.x] = best; si[threadIdx.x] = bi; __syncthreads();
    for (int o = 128; o > 0; o >>= 1) {
        if ((int)threadIdx.x < o) {
            float ov = sv[threadIdx.x + o]; int oi = si[threadIdx.x + o];
            if (ov > sv[threadIdx.x] || (ov == sv[threadIdx.x] && oi < si[threadIdx.x])) {
                sv[threadIdx.x] = ov; si[threadIdx.x] = oi;
            }
        }
        __syncthreads();
    }
    if (threadIdx.x == 0) { pmax[blockIdx.x] = sv[0]; pidx[blockIdx.x] = si[0]; }
}

__global__ void k_red_max2(const float* pmax, const int* pidx, float* cells) {
    __shared__ float sv[256];
    __shared__ int   si[256];
    float best = -INFINITY; int bi = 0x7fffffff;
    for (int i = threadIdx.x; i < RBLK; i += 256) {
        float v = pmax[i]; int oi = pidx[i];
        if (v > best || (v == best && oi < bi)) { best = v; bi = oi; }
    }
    sv[threadIdx.x] = best; si[threadIdx.x] = bi; __syncthreads();
    for (int o = 128; o > 0; o >>= 1) {
        if ((int)threadIdx.x < o) {
            float ov = sv[threadIdx.x + o]; int oi = si[threadIdx.x + o];
            if (ov > sv[threadIdx.x] || (ov == sv[threadIdx.x] && oi < si[threadIdx.x])) {
                sv[threadIdx.x] = ov; si[threadIdx.x] = oi;
            }
        }
        __syncthreads();
    }
    if (threadIdx.x == 0) { cells[0] = sv[0]; ((int*)cells)[2] = si[0]; }
}

// ---------------------------------------------------------------- masked sum(exp) (2-stage)
__global__ void k_red_sum(const float* __restrict__ scores, const float* __restrict__ x,
                          const int* __restrict__ dsi, const float* cells, float* psum) {
    __shared__ float sv[256];
    float gmax = cells[0];
    float acc = 0.0f;
    for (int i = blockIdx.x * blockDim.x + threadIdx.x; i < NM; i += gridDim.x * blockDim.x) {
        if (mask_of(x, dsi, i)) acc += expf(scores[i] - gmax);
    }
    sv[threadIdx.x] = acc; __syncthreads();
    for (int o = 128; o > 0; o >>= 1) {
        if ((int)threadIdx.x < o) sv[threadIdx.x] += sv[threadIdx.x + o];
        __syncthreads();
    }
    if (threadIdx.x == 0) psum[blockIdx.x] = sv[0];
}

__global__ void k_red_sum2(const float* psum, float* cells) {
    __shared__ float sv[256];
    float acc = 0.0f;
    for (int i = threadIdx.x; i < RBLK; i += 256) acc += psum[i];
    sv[threadIdx.x] = acc; __syncthreads();
    for (int o = 128; o > 0; o >>= 1) {
        if ((int)threadIdx.x < o) sv[threadIdx.x] += sv[threadIdx.x + o];
        __syncthreads();
    }
    if (threadIdx.x == 0) cells[1] = sv[0];
}

// ---------------------------------------------------------------- probs + selected
__global__ void k_final(const float* __restrict__ scores, const float* __restrict__ x,
                        const int* __restrict__ dsi, const float* cells, float* out) {
    int m = blockIdx.x * blockDim.x + threadIdx.x;
    if (m < NM) {
        float gmax = cells[0], Z = cells[1];
        out[m] = mask_of(x, dsi, m) ? expf(scores[m] - gmax) / Z : 0.0f;
    } else if (m == NM) {
        out[NM] = (float)((const int*)cells)[2];
    }
}

// ================================================================ launcher
extern "C" void kernel_launch(void* const* d_in, const int* in_sizes, int n_in,
                              void* d_out, int out_size, void* d_ws, size_t ws_size,
                              hipStream_t stream) {
    const float* x   = (const float*)d_in[0];
    const int*   ei  = (const int*)  d_in[1];
    const int*   dsi = (const int*)  d_in[2];
    const float* W1  = (const float*)d_in[3];
    const float* b1  = (const float*)d_in[4];
    const float* W2  = (const float*)d_in[5];
    const float* b2  = (const float*)d_in[6];
    const float* M1  = (const float*)d_in[7];
    const float* mb1 = (const float*)d_in[8];
    const float* M2  = (const float*)d_in[9];
    const float* mb2 = (const float*)d_in[10];
    const float* M3  = (const float*)d_in[11];
    const float* mb3 = (const float*)d_in[12];
    float* out = (float*)d_out;

    char* w = (char*)d_ws;
    size_t off = 0;
    auto carve = [&](size_t bytes) { size_t c = off; off += (bytes + 255) & ~(size_t)255; return c; };
    float*  dinv   = (float*) (w + carve((size_t)NP * 4));
    float*  bufA   = (float*) (w + carve((size_t)NP * HDIM * 4));
    float*  bufB   = (float*) (w + carve((size_t)NP * HDIM * 4));
    __bf16* bufC   = (__bf16*)(w + carve((size_t)NP * HDIM * 2));
    __bf16* w2t    = (__bf16*)(w + carve(64 * 64 * 2));
    __bf16* m1t    = (__bf16*)(w + carve(192 * 64 * 2));
    __bf16* m2t    = (__bf16*)(w + carve(64 * 64 * 2));
    float*  scores = (float*) (w + carve((size_t)NM * 4));
    float*  pmax   = (float*) (w + carve(RBLK * 4));
    int*    pidx   = (int*)   (w + carve(RBLK * 4));
    float*  psum   = (float*) (w + carve(RBLK * 4));
    float*  cells  = (float*) (w + carve(256));

    const int T = 256;
    auto cdiv = [](long long a, long long b) { return (int)((a + b - 1) / b); };

    // degree/norm + weight staging
    k_init_deg   <<<cdiv(NP, T), T, 0, stream>>>(dinv);
    k_cvt_weights<<<cdiv(4096 + 12288 + 4096, T), T, 0, stream>>>(W2, M1, M2, w2t, m1t, m2t);
    k_degree     <<<cdiv(EDGES, T), T, 0, stream>>>(ei, dinv);
    k_rsqrt      <<<cdiv(N_NODES, T), T, 0, stream>>>(dinv);

    // ---- GCN layer 1 (hW -> bufA, fused agg seed -> bufB)
    k_gemm1   <<<cdiv((long long)N_NODES * HDIM, T), T, 0, stream>>>(x, W1, dinv, b1, bufA, bufB);
    k_edges   <<<cdiv((long long)EDGES * 32, T), T, 0, stream>>>(ei, dinv, bufA, bufB);
    k_relu_cvt<<<cdiv((long long)NP * HDIM, T), T, 0, stream>>>(bufB, bufC);

    // ---- GCN layer 2 (WMMA, fused agg seed)
    int tilesConv = (NP / 16) * 4;
    k_wmma_conv2<<<cdiv((long long)tilesConv * 32, T), T, 0, stream>>>(
                    bufC, w2t, dinv, b2, bufA, bufB);
    k_edges   <<<cdiv((long long)EDGES * 32, T), T, 0, stream>>>(ei, dinv, bufA, bufB);
    k_relu_cvt<<<cdiv((long long)NP * HDIM, T), T, 0, stream>>>(bufB, bufC);

    // ---- MLP (WMMA): z1 = relu([h,dag,gl] @ M1 + mb1) -> bf16 in bufA
    int tilesMlp = (NM / 16) * 4;
    __bf16* z1 = (__bf16*)bufA;
    k_wmma_mlp1<<<cdiv((long long)tilesMlp * 32, T), T, 0, stream>>>(bufC, m1t, dsi, mb1, z1);
    // z2 + M3 dot fused in registers -> scores
    k_wmma_mlp2<<<cdiv((long long)(NM / 16) * 32, T), T, 0, stream>>>(
                   z1, m2t, mb2, M3, mb3, scores);

    // ---- masked softmax + argmax
    k_red_max <<<RBLK, T, 0, stream>>>(scores, x, dsi, pmax, pidx);
    k_red_max2<<<1,    T, 0, stream>>>(pmax, pidx, cells);
    k_red_sum <<<RBLK, T, 0, stream>>>(scores, x, dsi, cells, psum);
    k_red_sum2<<<1,    T, 0, stream>>>(psum, cells);
    k_final   <<<cdiv(NM + 1, T), T, 0, stream>>>(scores, x, dsi, cells, out);

    (void)in_sizes; (void)n_in; (void)out_size; (void)ws_size;
}
